// GAT_66305705116167
// MI455X (gfx1250) — 16-bit and 8-bit matrix operand order repaired and checked by reading, compiled, not run on hardware
//
#include <hip/hip_runtime.h>
#include <cmath>

#define BB 2
#define NN 2048
#define NFEAT 256
#define NHID 64
#define NHEADS 8
#define NCLASS 16
#define ALPHA 0.2f
#define NEGINF -1.0e12f

typedef __attribute__((ext_vector_type(16))) _Float16 v16h;
typedef __attribute__((ext_vector_type(8)))  float    v8f;
typedef __attribute__((ext_vector_type(8)))  int      v8i;

__global__ void k_cvt_x(const float* __restrict__ x, _Float16* __restrict__ xf) {
    int t = blockIdx.x * blockDim.x + threadIdx.x;
    if (t < BB * NN * NFEAT) xf[t] = (_Float16)x[t];
}

__global__ void k_cvt_W(const float* __restrict__ Ws, _Float16* __restrict__ WT) {
    int t = blockIdx.x * blockDim.x + threadIdx.x;
    if (t >= NHEADS * NFEAT * NHID) return;
    int h = t / (NFEAT * NHID);
    int r = t % (NFEAT * NHID);
    int k = r / NHID, o = r % NHID;
    WT[((size_t)h * NHID + o) * NFEAT + k] = (_Float16)Ws[t];
}

__global__ void k_cvt_Wo(const float* __restrict__ Wo, _Float16* __restrict__ WoT) {
    int t = blockIdx.x * blockDim.x + threadIdx.x;
    if (t >= NHEADS * NHID * NCLASS) return;
    int k = t / NCLASS, c = t % NCLASS;
    WoT[(size_t)c * (NHEADS * NHID) + k] = (_Float16)Wo[t];
}

__global__ void k_pack_adj(const float* __restrict__ adj,
                           unsigned char* __restrict__ adjb,
                           float* __restrict__ adjw, float w1) {
    int t = blockIdx.x * blockDim.x + threadIdx.x;
    if (t >= BB * NN * NN) return;
    float a = adj[t];
    adjb[t] = (a != 0.0f) ? 1 : 0;
    int i = (t / NN) % NN, j = t % NN;
    adjw[t] = (i == j) ? 1.0f : w1 * a;
}

__global__ __launch_bounds__(32)
void k_graph_step(const unsigned char* __restrict__ prev,
                  const unsigned char* __restrict__ adjb,
                  unsigned char* __restrict__ next,
                  float* __restrict__ adjw, float w) {
    const int lane = threadIdx.x, r = lane & 15, half = lane >> 4;
    const int ti = blockIdx.x * 32, tj = blockIdx.y * 32, b = blockIdx.z;
    const size_t base = (size_t)b * NN * NN;
    const unsigned char* arow0 = prev + base + (size_t)(ti + r) * NN;
    const unsigned char* arow1 = prev + base + (size_t)(ti + 16 + r) * NN;
    const unsigned char* brow0 = adjb + base + (size_t)(tj + r) * NN;
    const unsigned char* brow1 = adjb + base + (size_t)(tj + 16 + r) * NN;
    v8i c00 = {}, c01 = {}, c10 = {}, c11 = {};
#pragma unroll 2
    for (int k = 0; k < NN; k += 64) {
        union { v8i v; unsigned long long q[4]; } A0, A1;
        {
            const unsigned char* ap = arow0 + k + half * 16;
            A0.q[0] = *(const unsigned long long*)(ap + 0);
            A0.q[1] = *(const unsigned long long*)(ap + 8);
            A0.q[2] = *(const unsigned long long*)(ap + 32);
            A0.q[3] = *(const unsigned long long*)(ap + 40);
        }
        {
            const unsigned char* ap = arow1 + k + half * 16;
            A1.q[0] = *(const unsigned long long*)(ap + 0);
            A1.q[1] = *(const unsigned long long*)(ap + 8);
            A1.q[2] = *(const unsigned long long*)(ap + 32);
            A1.q[3] = *(const unsigned long long*)(ap + 40);
        }
        union { v8i v; uint4 u[2]; } B0, B1;
        {
            const unsigned char* bp = brow0 + k + half * 16;
            B0.u[0] = *(const uint4*)(bp + 0);
            B0.u[1] = *(const uint4*)(bp + 32);
        }
        {
            const unsigned char* bp = brow1 + k + half * 16;
            B1.u[0] = *(const uint4*)(bp + 0);
            B1.u[1] = *(const uint4*)(bp + 32);
        }
        c00 = __builtin_amdgcn_wmma_i32_16x16x64_iu8(false, A0.v, false, B0.v, c00,
                                                     false, false);
        c01 = __builtin_amdgcn_wmma_i32_16x16x64_iu8(false, A0.v, false, B1.v, c01,
                                                     false, false);
        c10 = __builtin_amdgcn_wmma_i32_16x16x64_iu8(false, A1.v, false, B0.v, c10,
                                                     false, false);
        c11 = __builtin_amdgcn_wmma_i32_16x16x64_iu8(false, A1.v, false, B1.v, c11,
                                                     false, false);
    }
    const v8i* cs[2][2] = {{&c00, &c01}, {&c10, &c11}};
#pragma unroll
    for (int ii = 0; ii < 2; ++ii)
#pragma unroll
        for (int jj = 0; jj < 2; ++jj) {
            const v8i& c = *cs[ii][jj];
#pragma unroll
            for (int v = 0; v < 8; ++v) {
                int i = ti + ii * 16 + v + 8 * half;
                int j = tj + jj * 16 + r;
                size_t idx = base + (size_t)i * NN + j;
                unsigned char er = (c[v] > 0) ? 1 : 0;
                unsigned char pv = prev[idx];
                next[idx] = er;
                if (er && !pv) adjw[idx] = w;
            }
        }
}

__global__ __launch_bounds__(32)
void k_hgemm(const _Float16* __restrict__ xf,
             const _Float16* __restrict__ WT,
             _Float16* __restrict__ hT) {
    const int lane = threadIdx.x, r = lane & 15, half = lane >> 4;
    const int ti = blockIdx.x * 16;
    const int bh = blockIdx.y, b = bh >> 3, h = bh & 7;
    const _Float16* arow = xf + ((size_t)b * NN + ti + r) * NFEAT;
    v8f acc[4] = {};
#pragma unroll
    for (int k = 0; k < NFEAT; k += 32) {
        union { v16h v; uint4 u[2]; } A;
        A.u[0] = *(const uint4*)(arow + k + half * 16);
        A.u[1] = *(const uint4*)(arow + k + half * 16 + 8);
#pragma unroll
        for (int t = 0; t < 4; ++t) {
            const _Float16* bp =
                WT + ((size_t)h * NHID + t * 16 + r) * NFEAT + k + half * 16;
            v16h Bv = *(const v16h*)bp;
            acc[t] = __builtin_amdgcn_wmma_f32_16x16x32_f16(
                false, A.v, false, Bv, (short)0, acc[t], false, false);
        }
    }
#pragma unroll
    for (int t = 0; t < 4; ++t)
#pragma unroll
        for (int v = 0; v < 8; ++v) {
            int o = t * 16 + r, i = ti + v + 8 * half;
            hT[(((size_t)b * NHEADS + h) * NHID + o) * NN + i] = (_Float16)acc[t][v];
        }
}

__global__ void k_f12(const _Float16* __restrict__ hT, const float* __restrict__ Aa,
                      float* __restrict__ f1, float* __restrict__ f2,
                      int BH, int Hh, int D) {
    int t = blockIdx.x * blockDim.x + threadIdx.x;
    if (t >= BH * NN) return;
    int bh = t / NN, n = t % NN, h = bh % Hh;
    const _Float16* hp = hT + (size_t)bh * D * NN + n;
    const float* a1 = Aa + (size_t)h * 2 * D;
    float s1 = 0.f, s2 = 0.f;
    for (int o = 0; o < D; ++o) {
        float hv = (float)hp[(size_t)o * NN];
        s1 += hv * a1[o];
        s2 += hv * a1[D + o];
    }
    f1[t] = s1; f2[t] = s2;
}

template <int NT, bool FINAL>
__global__ __launch_bounds__(32)
void k_attn(const float* __restrict__ adjw, const float* __restrict__ f1,
            const float* __restrict__ f2, const _Float16* __restrict__ hT,
            _Float16* __restrict__ xh16, float* __restrict__ outf, int Hh) {
    const int lane = threadIdx.x, r = lane & 15, half = lane >> 4;
    const int ti = blockIdx.x * 16;
    const int bh = blockIdx.y, b = bh / Hh;
    const float* wrow  = adjw + ((size_t)b * NN + ti + r) * NN;
    const float* f2row = f2 + (size_t)bh * NN;
    const float f1v = f1[(size_t)bh * NN + ti + r];
    const _Float16* hbase = hT + (size_t)bh * (NT * 16) * NN;

    float m = -1e30f, l = 0.f;
    v8f acc[NT];
#pragma unroll
    for (int t = 0; t < NT; ++t) acc[t] = (v8f){};

    for (int j = 0; j < NN; j += 32) {
        float s[16], lmax = -1e30f;
#pragma unroll
        for (int g = 0; g < 2; ++g) {
            const float* wp = wrow + j + half * 16 + g * 8;
            const float* fp = f2row + j + half * 16 + g * 8;
            float4 w0 = *(const float4*)(wp), w1 = *(const float4*)(wp + 4);
            float4 q0 = *(const float4*)(fp), q1 = *(const float4*)(fp + 4);
            float wv[8] = {w0.x, w0.y, w0.z, w0.w, w1.x, w1.y, w1.z, w1.w};
            float qv[8] = {q0.x, q0.y, q0.z, q0.w, q1.x, q1.y, q1.z, q1.w};
#pragma unroll
            for (int e = 0; e < 8; ++e) {
                float ee = f1v + qv[e];
                ee = (ee > 0.f) ? ee : ALPHA * ee;
                float sv = (wv[e] > 0.f) ? wv[e] * ee : NEGINF;
                s[g * 8 + e] = sv;
                lmax = fmaxf(lmax, sv);
            }
        }
        lmax = fmaxf(lmax, __shfl_xor(lmax, 16, 32));
        float mnew = fmaxf(m, lmax);
        float sc = __expf(m - mnew);
        float lsum = 0.f;
        union { v16h v; _Float16 e[16]; } P;
#pragma unroll
        for (int e = 0; e < 16; ++e) {
            float pv = __expf(s[e] - mnew);
            lsum += pv;
            P.e[e] = (_Float16)pv;
        }
        lsum += __shfl_xor(lsum, 16, 32);
        l = l * sc + lsum;
        m = mnew;
        float scv[8];
#pragma unroll
        for (int v = 0; v < 8; ++v) scv[v] = __shfl(sc, v + 8 * half, 32);
#pragma unroll
        for (int t = 0; t < NT; ++t)
#pragma unroll
            for (int v = 0; v < 8; ++v) acc[t][v] *= scv[v];
#pragma unroll
        for (int t = 0; t < NT; ++t) {
            const _Float16* bp = hbase + (size_t)(t * 16 + r) * NN + j + half * 16;
            v16h Bv = *(const v16h*)bp;
            acc[t] = __builtin_amdgcn_wmma_f32_16x16x32_f16(
                false, P.v, false, Bv, (short)0, acc[t], false, false);
        }
    }
    float lv[8];
#pragma unroll
    for (int v = 0; v < 8; ++v) lv[v] = __shfl(l, v + 8 * half, 32);
#pragma unroll
    for (int t = 0; t < NT; ++t)
#pragma unroll
        for (int v = 0; v < 8; ++v) {
            float o = acc[t][v] / lv[v];
            o = (o > 0.f) ? o : (__expf(o) - 1.0f);
            int i = ti + v + 8 * half;
            if constexpr (FINAL) {
                outf[((size_t)b * NN + i) * NCLASS + t * 16 + r] = o;
            } else {
                int h = bh % Hh;
                xh16[((size_t)b * NN + i) * (NHEADS * NHID) + h * NHID + t * 16 + r] =
                    (_Float16)o;
            }
        }
}

__global__ __launch_bounds__(32)
void k_outgemm(const _Float16* __restrict__ xh16, const _Float16* __restrict__ WoT,
               _Float16* __restrict__ houtT) {
    const int lane = threadIdx.x, r = lane & 15, half = lane >> 4;
    const int ti = blockIdx.x * 16, b = blockIdx.y;
    const int K = NHEADS * NHID;
    const _Float16* arow = xh16 + ((size_t)b * NN + ti + r) * K;
    v8f acc = {};
#pragma unroll 4
    for (int k = 0; k < K; k += 32) {
        union { v16h v; uint4 u[2]; } A;
        A.u[0] = *(const uint4*)(arow + k + half * 16);
        A.u[1] = *(const uint4*)(arow + k + half * 16 + 8);
        v16h Bv = *(const v16h*)(WoT + (size_t)r * K + k + half * 16);
        acc = __builtin_amdgcn_wmma_f32_16x16x32_f16(false, A.v, false, Bv,
                                                     (short)0, acc, false, false);
    }
#pragma unroll
    for (int v = 0; v < 8; ++v) {
        int i = ti + v + 8 * half;
        houtT[((size_t)b * NCLASS + r) * NN + i] = (_Float16)acc[v];
    }
}

extern "C" void kernel_launch(void* const* d_in, const int* in_sizes, int n_in,
                              void* d_out, int out_size, void* d_ws, size_t ws_size,
                              hipStream_t stream) {
    const float* x     = (const float*)d_in[0];
    const float* adj   = (const float*)d_in[1];
    const float* Ws    = (const float*)d_in[2];
    const float* As    = (const float*)d_in[3];
    const float* W_out = (const float*)d_in[4];
    const float* a_out = (const float*)d_in[5];
    (void)in_sizes; (void)n_in; (void)out_size; (void)ws_size;
    float* out = (float*)d_out;

    char* p = (char*)d_ws;
    auto alloc = [&](size_t bytes) -> void* {
        void* r = (void*)p;
        p += (bytes + 255) & ~(size_t)255;
        return r;
    };
    unsigned char* adjb = (unsigned char*)alloc((size_t)BB * NN * NN);
    unsigned char* nxt1 = (unsigned char*)alloc((size_t)BB * NN * NN);
    unsigned char* nxt2 = (unsigned char*)alloc((size_t)BB * NN * NN);
    float*     adjw  = (float*)alloc((size_t)BB * NN * NN * 4);
    _Float16*  xf    = (_Float16*)alloc((size_t)BB * NN * NFEAT * 2);
    _Float16*  WT    = (_Float16*)alloc((size_t)NHEADS * NHID * NFEAT * 2);
    _Float16*  WoT   = (_Float16*)alloc((size_t)NCLASS * NHEADS * NHID * 2);
    _Float16*  hT    = (_Float16*)alloc((size_t)BB * NHEADS * NHID * NN * 2);
    float*     f1    = (float*)alloc((size_t)BB * NHEADS * NN * 4);
    float*     f2    = (float*)alloc((size_t)BB * NHEADS * NN * 4);
    _Float16*  xh16  = (_Float16*)alloc((size_t)BB * NN * NHEADS * NHID * 2);
    _Float16*  houtT = (_Float16*)alloc((size_t)BB * NCLASS * NN * 2);
    float*     f1o   = (float*)alloc((size_t)BB * NN * 4);
    float*     f2o   = (float*)alloc((size_t)BB * NN * 4);

    const float sigma2 = 4.5f;
    const float w1 = expf(-1.0f / sigma2);
    const float w2 = expf(-4.0f / sigma2);
    const float w3 = expf(-9.0f / sigma2);

    k_cvt_x<<<(BB * NN * NFEAT + 255) / 256, 256, 0, stream>>>(x, xf);
    k_cvt_W<<<(NHEADS * NFEAT * NHID + 255) / 256, 256, 0, stream>>>(Ws, WT);
    k_cvt_Wo<<<(NHEADS * NHID * NCLASS + 255) / 256, 256, 0, stream>>>(W_out, WoT);
    k_pack_adj<<<((size_t)BB * NN * NN + 255) / 256, 256, 0, stream>>>(adj, adjb,
                                                                       adjw, w1);

    dim3 gGraph(NN / 32, NN / 32, BB);
    k_graph_step<<<gGraph, 32, 0, stream>>>(adjb, adjb, nxt1, adjw, w2);
    k_graph_step<<<gGraph, 32, 0, stream>>>(nxt1, adjb, nxt2, adjw, w3);

    k_hgemm<<<dim3(NN / 16, BB * NHEADS), 32, 0, stream>>>(xf, WT, hT);
    k_f12<<<(BB * NHEADS * NN + 255) / 256, 256, 0, stream>>>(hT, As, f1, f2,
                                                              BB * NHEADS, NHEADS,
                                                              NHID);
    k_attn<4, false><<<dim3(NN / 16, BB * NHEADS), 32, 0, stream>>>(
        adjw, f1, f2, hT, xh16, nullptr, NHEADS);

    k_outgemm<<<dim3(NN / 16, BB), 32, 0, stream>>>(xh16, WoT, houtT);
    k_f12<<<(BB * NN + 255) / 256, 256, 0, stream>>>(houtT, a_out, f1o, f2o, BB, 1,
                                                     NCLASS);
    k_attn<1, true><<<dim3(NN / 16, BB), 32, 0, stream>>>(adjw, f1o, f2o, houtT,
                                                          nullptr, out, 1);
}
